// TeluguAntiRepetitionLoss_28759101014346
// MI455X (gfx1250) — matrix operand order repaired and checked by reading, compiled, not run on hardware
//
#include <hip/hip_runtime.h>
#include <hip/hip_bf16.h>

// ---------------------------------------------------------------------------
// TeluguAntiRepetitionLoss for MI455X (gfx1250, wave32, WMMA)
//
// Shapes: logits (B=4, T=1024, V=32000) fp32, input_ids (4,1024) int32.
// Output: 4 fp32 scalars [l_token, l_con, l_ng, total].
//
//   K1: one-pass online softmax stats (rowmax, rowsum). Single 512MB stream
//       read == HBM roofline (~22us @ 23.3 TB/s). Never materialize probs.
//   K2: token/consec losses via sparse gathers + per-block partials.
//   K3: ngram dup einsum (4 x 1022^3 MACs) with v_wmma_f32_16x16x32_f16.
//       0/1 fragments built BRANCHLESSLY from sentinel-padded LDS arrays,
//       int4 (ds_load_b128) reads; EXEC stays full, no per-element waits.
//   K4: deterministic fixed-order reduction + final scalars (no atomics).
// ---------------------------------------------------------------------------

typedef __attribute__((ext_vector_type(16))) _Float16 v16h;
typedef __attribute__((ext_vector_type(8)))  float    v8f;

#define BB 4
#define TT 1024
#define VV 32000
#define WINDOW 20
#define TM 1022          // Tm = T - (NGRAM_N-1)
#define NTILE 64         // 1024/16 tiles per dim (Tm padded to 1024)

// ---------------- Kernel 1: per-row online max / sum-exp --------------------
__global__ void row_stats_kernel(const float* __restrict__ logits,
                                 float* __restrict__ rowmax,
                                 float* __restrict__ rowsum) {
  const int row = blockIdx.x;                  // 0 .. B*T-1
  const int tid = threadIdx.x;                 // 256 threads = 8 waves
  const float4* x4 = (const float4*)(logits + (size_t)row * VV);
  const int n4 = VV / 4;                       // 8000, exact

  float m = -INFINITY, s = 0.0f;
  for (int i = tid; i < n4; i += 256) {
    float4 v = x4[i];
    float lm = fmaxf(fmaxf(v.x, v.y), fmaxf(v.z, v.w));
    if (lm > m) { s *= __expf(m - lm); m = lm; }
    s += __expf(v.x - m) + __expf(v.y - m) + __expf(v.z - m) + __expf(v.w - m);
  }

  __shared__ float sm[256], ss[256];
  sm[tid] = m; ss[tid] = s;
  __syncthreads();
  for (int off = 128; off > 0; off >>= 1) {
    if (tid < off) {
      float m2 = sm[tid + off], s2 = ss[tid + off];
      float mm = fmaxf(sm[tid], m2);
      ss[tid] = ss[tid] * __expf(sm[tid] - mm) + s2 * __expf(m2 - mm);
      sm[tid] = mm;
    }
    __syncthreads();
  }
  if (tid == 0) { rowmax[row] = sm[0]; rowsum[row] = ss[0]; }
}

// ---------------- Kernel 2: token + consec losses ---------------------------
__global__ void token_consec_kernel(const float* __restrict__ logits,
                                    const int* __restrict__ ids,
                                    const float* __restrict__ rowmax,
                                    const float* __restrict__ rowsum,
                                    float* __restrict__ bpart /* 3 per block */) {
  const int tid = threadIdx.x;
  const int g = blockIdx.x * 256 + tid;        // 0 .. 4095
  float tok = 0.0f, con = 0.0f, cnt = 0.0f;

  if (g < BB * TT) {
    const int b = g / TT, t = g % TT;
    const float* xrow = logits + (size_t)g * VV;
    const float m = rowmax[g];
    const float inv = 1.0f / rowsum[g];

    int p0 = t - WINDOW; if (p0 < 0) p0 = 0;
    for (int p = p0; p < t; ++p) {
      int id = ids[b * TT + p];
      if (id != 0) {
        float prob = __expf(xrow[id] - m) * inv;
        float w = __expf(-(float)(t - p - 1) * 0.2f);
        tok += prob * w;
      }
    }
    if (t >= 1) {
      int id = ids[b * TT + t - 1];
      if (id != 0) {
        con = __expf(xrow[id] - m) * inv;
        cnt = 1.0f;
      }
    }
  }

  __shared__ float sh[3 * 256];
  sh[tid] = tok; sh[256 + tid] = con; sh[512 + tid] = cnt;
  __syncthreads();
  for (int off = 128; off > 0; off >>= 1) {
    if (tid < off) {
      sh[tid]       += sh[tid + off];
      sh[256 + tid] += sh[256 + tid + off];
      sh[512 + tid] += sh[512 + tid + off];
    }
    __syncthreads();
  }
  if (tid == 0) {
    bpart[3 * blockIdx.x + 0] = sh[0];
    bpart[3 * blockIdx.x + 1] = sh[256];
    bpart[3 * blockIdx.x + 2] = sh[512];
  }
}

// ---------------- Kernel 3: ngram loss via WMMA -----------------------------
// dup[b,u,i] = sum_j M[b,u,j] * E[b,j,i]   (0/1 matrices, exact in f16/f32)
// One wave per 16x16 tile; K in steps of 32 via v_wmma_f32_16x16x32_f16.
__device__ __forceinline__ _Float16 sel01(int c) {
  return c ? (_Float16)1.0f : (_Float16)0.0f;
}

__global__ void
__launch_bounds__(32)
ngram_wmma_kernel(const float* __restrict__ logits,
                  const int* __restrict__ ids,
                  const float* __restrict__ rowmax,
                  const float* __restrict__ rowsum,
                  float* __restrict__ cpart /* 2 per block */) {
  const int b  = blockIdx.z;
  const int u0 = blockIdx.y * 16;     // dup row tile
  const int i0 = blockIdx.x * 16;     // dup col tile
  const int lane = threadIdx.x;       // 0..31 (wave32)
  const int hi = lane >> 4;           // lane group
  const int lp = lane & 15;

  // Sentinel-padded shifted copies of the id row:
  //   s_p0[j] = ids[j]            (P0)
  //   s_p1[j] = ids[j+1] if j<TM else SENT   (P1; kills j>=TM in A)
  //   s_c[j]  = ids[j+2] if j<TM else SENT2  (C)
  __shared__ int s_p0[TT], s_p1[TT], s_c[TT];
  for (int i = lane; i < TT; i += 32) s_p0[i] = ids[b * TT + i];
  __syncthreads();
  for (int i = lane; i < TT; i += 32) {
    s_p1[i] = (i < TM) ? s_p0[i + 1] : 0x7fffffff;
    s_c[i]  = (i < TM) ? s_p0[i + 2] : 0x7ffffffe;
  }
  __syncthreads();

  // Loop invariants. Invalid lanes get -1/-2: never equal to any id/sentinel.
  const int  uA  = u0 + lp;                    // A row (M = lp, both groups)
  const bool uok = (uA < TM);
  const int  p0u = uok ? s_p0[uA] : -1;
  const int  p1u = uok ? s_p1[uA] : -2;
  const int  iB  = i0 + lp;                    // B col (N = lp, both groups)
  const int  cI  = (iB < TM) ? s_c[iB] : -1;

  const int4* p04 = (const int4*)s_p0;
  const int4* p14 = (const int4*)s_p1;
  const int4* c4  = (const int4*)s_c;

  v8f acc = {};
#pragma unroll 2
  for (int j0 = 0; j0 < 1024; j0 += 32) {
    // A elems h=0..15 -> jA = j0+hi*16+h
    // B elems h=0..15 -> jB = j0+hi*16+h
    const int bA1 = (j0 + hi * 16) >> 2;       // int4 index, aligned
    const int bA2 = (j0 + hi * 16 + 8) >> 2;
    const int bB  = (j0 + hi * 16) >> 2;
    int4 a0 = p04[bA1], a1 = p04[bA1 + 1];
    int4 q0 = p14[bA1], q1 = p14[bA1 + 1];
    int4 a2 = p04[bA2], a3 = p04[bA2 + 1];
    int4 q2 = p14[bA2], q3 = p14[bA2 + 1];
    int4 c0 = c4[bB], c1 = c4[bB + 1], c2 = c4[bB + 2], c3 = c4[bB + 3];

    v16h a, bf;
    a[0]  = sel01((a0.x == p0u) & (q0.x == p1u));
    a[1]  = sel01((a0.y == p0u) & (q0.y == p1u));
    a[2]  = sel01((a0.z == p0u) & (q0.z == p1u));
    a[3]  = sel01((a0.w == p0u) & (q0.w == p1u));
    a[4]  = sel01((a1.x == p0u) & (q1.x == p1u));
    a[5]  = sel01((a1.y == p0u) & (q1.y == p1u));
    a[6]  = sel01((a1.z == p0u) & (q1.z == p1u));
    a[7]  = sel01((a1.w == p0u) & (q1.w == p1u));
    a[8]  = sel01((a2.x == p0u) & (q2.x == p1u));
    a[9]  = sel01((a2.y == p0u) & (q2.y == p1u));
    a[10] = sel01((a2.z == p0u) & (q2.z == p1u));
    a[11] = sel01((a2.w == p0u) & (q2.w == p1u));
    a[12] = sel01((a3.x == p0u) & (q3.x == p1u));
    a[13] = sel01((a3.y == p0u) & (q3.y == p1u));
    a[14] = sel01((a3.z == p0u) & (q3.z == p1u));
    a[15] = sel01((a3.w == p0u) & (q3.w == p1u));

    const int lim = iB - (j0 + hi * 16);       // elem h valid iff h < lim
    bf[0]  = sel01((0  < lim) & (c0.x == cI));
    bf[1]  = sel01((1  < lim) & (c0.y == cI));
    bf[2]  = sel01((2  < lim) & (c0.z == cI));
    bf[3]  = sel01((3  < lim) & (c0.w == cI));
    bf[4]  = sel01((4  < lim) & (c1.x == cI));
    bf[5]  = sel01((5  < lim) & (c1.y == cI));
    bf[6]  = sel01((6  < lim) & (c1.z == cI));
    bf[7]  = sel01((7  < lim) & (c1.w == cI));
    bf[8]  = sel01((8  < lim) & (c2.x == cI));
    bf[9]  = sel01((9  < lim) & (c2.y == cI));
    bf[10] = sel01((10 < lim) & (c2.z == cI));
    bf[11] = sel01((11 < lim) & (c2.w == cI));
    bf[12] = sel01((12 < lim) & (c3.x == cI));
    bf[13] = sel01((13 < lim) & (c3.y == cI));
    bf[14] = sel01((14 < lim) & (c3.z == cI));
    bf[15] = sel01((15 < lim) & (c3.w == cI));

    acc = __builtin_amdgcn_wmma_f32_16x16x32_f16(
        false, a, false, bf, (short)0, acc, false, false);
  }

  // Readout: C/D layout — lane&15 = N, VGPR g -> M=g (lanes<16) / g+8.
  float gsum = 0.0f, gcnt = 0.0f;
#pragma unroll
  for (int g = 0; g < 8; ++g) {
    int u = u0 + g + hi * 8;
    int i = i0 + lp;
    if ((u < TM) & (i < TM)) {
      int dup  = (acc[g] > 0.5f);
      int mval = (s_p0[u] == s_p0[i]) & (s_p1[u] == s_p1[i]);
      int ci   = s_c[i];
      if (mval & !dup & (ci != 0)) {
        size_t ridx = (size_t)b * TT + (u + 2);
        float m = rowmax[ridx];
        float s = rowsum[ridx];
        float x = logits[ridx * VV + ci];
        gsum += __expf(x - m) / s;
        gcnt += 1.0f;
      }
    }
  }

  for (int off = 16; off > 0; off >>= 1) {
    gsum += __shfl_xor(gsum, off, 32);
    gcnt += __shfl_xor(gcnt, off, 32);
  }
  if (lane == 0) {
    int bid = (blockIdx.z * NTILE + blockIdx.y) * NTILE + blockIdx.x;
    cpart[2 * bid + 0] = gsum;
    cpart[2 * bid + 1] = gcnt;
  }
}

// ---------------- Kernel 4: deterministic finalize --------------------------
__global__ void finalize_kernel(const float* __restrict__ bpart,
                                const float* __restrict__ cpart,
                                float* __restrict__ out) {
  const int tid = threadIdx.x;   // 256
  float tok = 0, con = 0, ccnt = 0, ngs = 0, ngc = 0;
  for (int i = tid; i < 16; i += 256) {
    tok  += bpart[3 * i + 0];
    con  += bpart[3 * i + 1];
    ccnt += bpart[3 * i + 2];
  }
  const int ncp = BB * NTILE * NTILE;          // 16384
  for (int i = tid; i < ncp; i += 256) {
    ngs += cpart[2 * i + 0];
    ngc += cpart[2 * i + 1];
  }
  __shared__ float sh[5 * 256];
  sh[tid] = tok; sh[256 + tid] = con; sh[512 + tid] = ccnt;
  sh[768 + tid] = ngs; sh[1024 + tid] = ngc;
  __syncthreads();
  for (int off = 128; off > 0; off >>= 1) {
    if (tid < off)
      for (int k = 0; k < 5; ++k)
        sh[k * 256 + tid] += sh[k * 256 + tid + off];
    __syncthreads();
  }
  if (tid == 0) {
    float token_loss  = sh[0]    / (4.0f * 1023.0f);      // B*(T-1)
    float consec_loss = sh[256]  / fmaxf(sh[512],  1.0f);
    float ngram_loss  = sh[768]  / fmaxf(sh[1024], 1.0f);
    float lt = 0.3f * token_loss;
    float lc = 0.8f * consec_loss;
    float ln = 0.5f * ngram_loss;
    out[0] = lt; out[1] = lc; out[2] = ln; out[3] = lt + lc + ln;
  }
}

// ---------------------------------------------------------------------------
extern "C" void kernel_launch(void* const* d_in, const int* in_sizes, int n_in,
                              void* d_out, int out_size, void* d_ws, size_t ws_size,
                              hipStream_t stream) {
  const float* logits = (const float*)d_in[0];
  const int*   ids    = (const int*)d_in[1];
  float* ws = (float*)d_ws;

  float* rowmax = ws;                 // 4096
  float* rowsum = ws + 4096;          // 4096
  float* bpart  = ws + 8192;          // 16 * 3
  float* cpart  = ws + 8192 + 48;     // 16384 * 2

  row_stats_kernel<<<BB * TT, 256, 0, stream>>>(logits, rowmax, rowsum);
  token_consec_kernel<<<(BB * TT) / 256, 256, 0, stream>>>(
      logits, ids, rowmax, rowsum, bpart);
  dim3 g3(NTILE, NTILE, BB);
  ngram_wmma_kernel<<<g3, 32, 0, stream>>>(logits, ids, rowmax, rowsum, cpart);
  finalize_kernel<<<1, 256, 0, stream>>>(bpart, cpart, (float*)d_out);
}
